// UnifiedSelfAttention_23502061044415
// MI455X (gfx1250) — matrix operand order repaired and checked by reading, not compiled, not run on hardware
//
#include <hip/hip_runtime.h>
#include <hip/hip_bf16.h>

typedef __bf16 bf16_t;
typedef bf16_t v16bf __attribute__((ext_vector_type(16)));
typedef float  v8f   __attribute__((ext_vector_type(8)));
typedef unsigned int u32;
typedef u32 u32x4 __attribute__((ext_vector_type(4)));
typedef unsigned short u16;

#define B_  2
#define L_  2048
#define E_  2048
#define H_  16
#define HD_ 128

__device__ __forceinline__ u16 f2bf(float f) {
  union { float f; u32 u; } v; v.f = f;
  u32 r = ((v.u >> 16) & 1u) + 0x7FFFu;   // round-to-nearest-even
  return (u16)((v.u + r) >> 16);
}

union FragU { v16bf v; u32x4 u[2]; };

// A-fragment: two contiguous 8-half (16B) runs
__device__ __forceinline__ v16bf frag2(const u16* p0, const u16* p1) {
  FragU f;
  f.u[0] = *(const u32x4*)p0;
  f.u[1] = *(const u32x4*)p1;
  return f.v;
}
// B-fragment: one contiguous 16-half (32B) run
__device__ __forceinline__ v16bf frag1(const u16* p) {
  FragU f;
  f.u[0] = ((const u32x4*)p)[0];
  f.u[1] = ((const u32x4*)p)[1];
  return f.v;
}

__device__ __forceinline__ v8f wmma_bf16(v16bf a, v16bf b, v8f c) {
  return __builtin_amdgcn_wmma_f32_16x16x32_bf16(false, a, false, b, (short)0, c,
                                                 false, false);
}

__device__ __forceinline__ v8f zero8() {
  v8f z;
#pragma unroll
  for (int e = 0; e < 8; ++e) z[e] = 0.0f;
  return z;
}

// LDS byte offset of a __shared__ object (generic-pointer low bits = LDS offset)
__device__ __forceinline__ u32 lds_off(const void* p) {
  return (u32)(uintptr_t)p;
}

// CDNA5 async copy: 16B global -> LDS, tracked with ASYNCcnt
__device__ __forceinline__ void async_b128(u32 lds_addr, const u16* g) {
  asm volatile("global_load_async_to_lds_b128 %0, %1, off"
               :: "v"(lds_addr), "v"(g) : "memory");
}
#define S_WAIT_ASYNCCNT(n) asm volatile("s_wait_asynccnt " #n ::: "memory")

// ---------------------------------------------------------------- fp32 -> bf16
__global__ void cvt_f32_bf16(const float* __restrict__ in, u16* __restrict__ out,
                             int n) {
  int i = blockIdx.x * blockDim.x + threadIdx.x;
  int stride = gridDim.x * blockDim.x;
  for (; i < n; i += stride) out[i] = f2bf(in[i]);
}

// ------------------------------------- fp32 [K,N] -> bf16 transposed [N,K]
__global__ __launch_bounds__(256)
void cvt_transpose_bf16(const float* __restrict__ in, u16* __restrict__ out,
                        int K, int N) {
  __shared__ u16 t[32][33];
  const int k0 = blockIdx.y * 32, n0 = blockIdx.x * 32;
  for (int i = threadIdx.x; i < 32 * 32; i += 256) {
    const int r = i >> 5, c = i & 31;
    t[r][c] = f2bf(in[(size_t)(k0 + r) * N + n0 + c]);
  }
  __syncthreads();
  for (int i = threadIdx.x; i < 32 * 32; i += 256) {
    const int r = i >> 5, c = i & 31;   // r: n-local, c: k-local
    out[(size_t)(n0 + r) * K + k0 + c] = t[c][r];
  }
}

// ------------------------------------------------- C[M,N]f32 = A * B^T + bias
// A: [M,K] bf16 row-major.  BT: [N,K] bf16 (pre-transposed weights).
// Block: 256 thr = 8 waves; tile 128x256, K-step 32, double-buffered async LDS.
// Wave grid 2(M) x 4(N); each wave 64x64 = 4x4 WMMA accumulators.
template <int N, int K>
__global__ __launch_bounds__(256)
void gemm_bf16(const u16* __restrict__ A, const u16* __restrict__ BT,
               const float* __restrict__ bias, float* __restrict__ C) {
  __shared__ u16 As[2][128 * 32];   // [m][k]
  __shared__ u16 Bs[2][256 * 32];   // [n][k]

  const int tid  = threadIdx.x;
  const int wave = tid >> 5;
  const int lane = tid & 31;
  const int lm = lane & 15, lg = lane >> 4;
  const int wm = wave >> 2;       // 0..1
  const int wn = wave & 3;        // 0..3
  const int m0 = blockIdx.y * 128;
  const int n0 = blockIdx.x * 256;

  v8f acc[4][4];
#pragma unroll
  for (int i = 0; i < 4; ++i)
#pragma unroll
    for (int j = 0; j < 4; ++j) acc[i][j] = zero8();

  const int arow = tid >> 1;            // 0..127
  const int aseg = (tid & 1) * 16;      // k segment (halves)
  const u16* Ag = A + (size_t)(m0 + arow) * K + aseg;
  const u16* Bg = BT + (size_t)(n0 + tid) * K;   // one 32-half row per thread

  auto stage = [&](int buf, int k0) {
    async_b128(lds_off(&As[buf][arow * 32 + aseg]),     Ag + k0);
    async_b128(lds_off(&As[buf][arow * 32 + aseg + 8]), Ag + k0 + 8);
    async_b128(lds_off(&Bs[buf][tid * 32]),      Bg + k0);
    async_b128(lds_off(&Bs[buf][tid * 32 + 8]),  Bg + k0 + 8);
    async_b128(lds_off(&Bs[buf][tid * 32 + 16]), Bg + k0 + 16);
    async_b128(lds_off(&Bs[buf][tid * 32 + 24]), Bg + k0 + 24);
  };

  stage(0, 0);
  constexpr int NT = K / 32;
  for (int t = 0; t < NT; ++t) {
    const int cur = t & 1;
    if (t + 1 < NT) {
      stage(cur ^ 1, (t + 1) * 32);
      S_WAIT_ASYNCCNT(6);     // 6 newest belong to next tile; current tile done
    } else {
      S_WAIT_ASYNCCNT(0);
    }
    __syncthreads();

    v16bf afr[4], bfr[4];
#pragma unroll
    for (int mt = 0; mt < 4; ++mt) {
      const u16* ap = &As[cur][(wm * 64 + mt * 16 + lm) * 32];
      afr[mt] = frag2(ap + 16 * lg, ap + 16 * lg + 8);
    }
#pragma unroll
    for (int nt = 0; nt < 4; ++nt) {
      const u16* bp = &Bs[cur][(wn * 64 + nt * 16 + lm) * 32];
      bfr[nt] = frag1(bp + 16 * lg);
    }
#pragma unroll
    for (int mt = 0; mt < 4; ++mt)
#pragma unroll
      for (int nt = 0; nt < 4; ++nt)
        acc[mt][nt] = wmma_bf16(afr[mt], bfr[nt], acc[mt][nt]);
    __syncthreads();
  }

#pragma unroll
  for (int mt = 0; mt < 4; ++mt)
#pragma unroll
    for (int nt = 0; nt < 4; ++nt) {
      const int col = n0 + wn * 64 + nt * 16 + lm;
      const float bv = bias[col];
      float* cp = C + (size_t)(m0 + wm * 64 + mt * 16 + 8 * lg) * N + col;
#pragma unroll
      for (int j = 0; j < 8; ++j) cp[(size_t)j * N] = acc[mt][nt][j] + bv;
    }
}

// ----------------------------------- RoPE + split + transpose to [B,H,L,HD] bf16
__global__ __launch_bounds__(256)
void rope_split(const float* __restrict__ qkv, const int* __restrict__ pos,
                u16* __restrict__ qt, u16* __restrict__ kt, u16* __restrict__ vt) {
  const int bl = blockIdx.x;            // 0 .. B*L-1
  const int b = bl / L_, l = bl % L_;
  const float p = (float)pos[bl];
  const float log_base = __logf(10000.0f);
  const float* row = qkv + (size_t)bl * (3 * E_);
  for (int idx = threadIdx.x; idx < E_; idx += blockDim.x) {
    const int h = idx / HD_, d = idx % HD_;
    const int pr = d & 63;
    const float freq = __expf(-(float)(2 * pr) * (1.0f / (float)HD_) * log_base);
    float s, c;
    __sincosf(p * freq, &s, &c);
    const size_t dst = ((size_t)(b * H_ + h) * L_ + l) * HD_ + d;

    const float q  = row[idx];
    const float qo = (d < 64) ? -row[idx + 64] : row[idx - 64];
    qt[dst] = f2bf(q * c + qo * s);

    const float* krow = row + E_;
    const float k  = krow[idx];
    const float ko = (d < 64) ? -krow[idx + 64] : krow[idx - 64];
    kt[dst] = f2bf(k * c + ko * s);

    vt[dst] = f2bf(row[2 * E_ + idx]);
  }
}

// -------------------------------------------------- flash attention (bf16 WMMA)
// grid: (L/128, B*H); block 256 = 8 waves; wave owns 16 q rows.
// Double-buffered 32-key chunks: K staged with async-to-LDS, V transposed.
__global__ __launch_bounds__(256)
void flash_attn(const u16* __restrict__ qt, const u16* __restrict__ kt,
                const u16* __restrict__ vt, u16* __restrict__ y) {
  __shared__ u16 Ks[2][32 * 128];    // [key][dim]
  __shared__ u16 VsT[2][128 * 32];   // [dim][key]
  __shared__ u16 Ps[8][16 * 32];     // per-wave probs [m][kcol]

  const int tid  = threadIdx.x;
  const int wave = tid >> 5, lane = tid & 31;
  const int lm = lane & 15, lg = lane >> 4;
  const int bh = blockIdx.y;
  const int b = bh / H_, h = bh % H_;
  const int q0 = blockIdx.x * 128 + wave * 16;
  const float scale = 0.08838834764831845f;  // 1/sqrt(128)

  // q fragments: 16 rows x HD, 4 chunks of K=32
  v16bf qf[4];
  {
    const u16* qrow = qt + ((size_t)bh * L_ + q0 + lm) * HD_;
#pragma unroll
    for (int ct = 0; ct < 4; ++ct)
      qf[ct] = frag2(qrow + ct * 32 + 16 * lg, qrow + ct * 32 + 16 * lg + 8);
  }

  v8f acc[8];
  float mi[8], li[8];
#pragma unroll
  for (int j = 0; j < 8; ++j) {
    acc[j] = zero8();
    mi[j] = -3.0e38f;
    li[j] = 0.0f;
  }

  const int krow_ld = tid >> 3;         // key 0..31
  const int kseg_ld = (tid & 7) * 16;   // dim segment

  auto stage = [&](int buf, int kb) {
    const u16* ksrc = kt + ((size_t)bh * L_ + kb + krow_ld) * HD_ + kseg_ld;
    async_b128(lds_off(&Ks[buf][krow_ld * 128 + kseg_ld]),     ksrc);
    async_b128(lds_off(&Ks[buf][krow_ld * 128 + kseg_ld + 8]), ksrc + 8);
    const u16* vsrc = vt + ((size_t)bh * L_ + kb + krow_ld) * HD_ + kseg_ld;
    u16 tmp[16];
    *(u32x4*)&tmp[0] = ((const u32x4*)vsrc)[0];
    *(u32x4*)&tmp[8] = ((const u32x4*)vsrc)[1];
#pragma unroll
    for (int i = 0; i < 16; ++i)
      VsT[buf][(kseg_ld + i) * 32 + krow_ld] = tmp[i];
  };

  stage(0, 0);
  for (int t = 0; t < L_ / 32; ++t) {
    const int cur = t & 1;
    if (t + 1 < L_ / 32) {
      stage(cur ^ 1, (t + 1) * 32);
      S_WAIT_ASYNCCNT(2);   // 2 newest = next chunk's K copies
    } else {
      S_WAIT_ASYNCCNT(0);
    }
    __syncthreads();

    // scores = q @ k^T : two 16x16 tiles (keys +0..15, +16..31)
    v8f s0 = zero8(), s1 = zero8();
#pragma unroll
    for (int ct = 0; ct < 4; ++ct) {
      v16bf b0 = frag1(&Ks[cur][lm * 128 + ct * 32 + 16 * lg]);
      v16bf b1 = frag1(&Ks[cur][(16 + lm) * 128 + ct * 32 + 16 * lg]);
      s0 = wmma_bf16(qf[ct], b0, s0);
      s1 = wmma_bf16(qf[ct], b1, s1);
    }

    // online softmax (row = j + 8*lg, cols striped over 16 lanes)
#pragma unroll
    for (int j = 0; j < 8; ++j) {
      float a0 = s0[j] * scale, a1 = s1[j] * scale;
      float t2 = fmaxf(a0, a1);
      t2 = fmaxf(t2, __shfl_xor(t2, 1));
      t2 = fmaxf(t2, __shfl_xor(t2, 2));
      t2 = fmaxf(t2, __shfl_xor(t2, 4));
      t2 = fmaxf(t2, __shfl_xor(t2, 8));
      const float mnew = fmaxf(mi[j], t2);
      const float sc = __expf(mi[j] - mnew);
      const float p0 = __expf(a0 - mnew);
      const float p1 = __expf(a1 - mnew);
      float rs = p0 + p1;
      rs += __shfl_xor(rs, 1);
      rs += __shfl_xor(rs, 2);
      rs += __shfl_xor(rs, 4);
      rs += __shfl_xor(rs, 8);
      li[j] = li[j] * sc + rs;
      mi[j] = mnew;
#pragma unroll
      for (int dt = 0; dt < 8; ++dt) acc[dt][j] *= sc;
      Ps[wave][(j + 8 * lg) * 32 + lm]      = f2bf(p0);
      Ps[wave][(j + 8 * lg) * 32 + 16 + lm] = f2bf(p1);
    }

    // acc += P @ V  (A = probs 16x32, B = V 32x16 per dim tile)
    const u16* prow = &Ps[wave][lm * 32];
    v16bf pf = frag2(prow + 16 * lg, prow + 16 * lg + 8);
#pragma unroll
    for (int dt = 0; dt < 8; ++dt) {
      v16bf vf = frag1(&VsT[cur][(dt * 16 + lm) * 32 + 16 * lg]);
      acc[dt] = wmma_bf16(pf, vf, acc[dt]);
    }
    __syncthreads();
  }

  // epilogue: normalize, write y[B,L,E] bf16
#pragma unroll
  for (int dt = 0; dt < 8; ++dt)
#pragma unroll
    for (int j = 0; j < 8; ++j) {
      const int row = q0 + j + 8 * lg;
      const float inv = 1.0f / li[j];
      y[((size_t)b * L_ + row) * E_ + h * HD_ + dt * 16 + lm] =
          f2bf(acc[dt][j] * inv);
    }
}

extern "C" void kernel_launch(void* const* d_in, const int* in_sizes, int n_in,
                              void* d_out, int out_size, void* d_ws, size_t ws_size,
                              hipStream_t stream) {
  const float* x   = (const float*)d_in[0];
  const int*   pos = (const int*)d_in[1];
  const float* Wa  = (const float*)d_in[2];
  const float* ba  = (const float*)d_in[3];
  const float* Wp  = (const float*)d_in[4];
  const float* bp  = (const float*)d_in[5];
  float* out = (float*)d_out;

  char* ws = (char*)d_ws;
  u16*   x_bf  = (u16*)(ws);                          // 16 MB
  u16*   waT   = (u16*)(ws + ((size_t)16 << 20));     // 24 MB  [3E, E]
  u16*   wpT   = (u16*)(ws + ((size_t)40 << 20));     //  8 MB  [E, E]
  float* qkv   = (float*)(ws + ((size_t)48 << 20));   // 96 MB
  u16*   q_t   = (u16*)(ws + ((size_t)144 << 20));    // 16 MB
  u16*   k_t   = (u16*)(ws + ((size_t)160 << 20));    // 16 MB
  u16*   v_t   = (u16*)(ws + ((size_t)176 << 20));    // 16 MB
  u16*   y_bf  = (u16*)(ws + ((size_t)192 << 20));    // 16 MB

  cvt_f32_bf16<<<2048, 256, 0, stream>>>(x, x_bf, B_ * L_ * E_);
  // W_attn [E, 3E] -> bf16 [3E, E];  W_proj [E, E] -> bf16 [E, E] transposed
  cvt_transpose_bf16<<<dim3(3 * E_ / 32, E_ / 32), 256, 0, stream>>>(Wa, waT,
                                                                     E_, 3 * E_);
  cvt_transpose_bf16<<<dim3(E_ / 32, E_ / 32), 256, 0, stream>>>(Wp, wpT,
                                                                 E_, E_);

  // QKV projection: [4096,2048] x [2048,6144]
  gemm_bf16<3 * E_, E_><<<dim3(3 * E_ / 256, (B_ * L_) / 128), 256, 0, stream>>>(
      x_bf, waT, ba, qkv);
  rope_split<<<B_ * L_, 256, 0, stream>>>(qkv, pos, q_t, k_t, v_t);
  flash_attn<<<dim3(L_ / 128, B_ * H_), 256, 0, stream>>>(q_t, k_t, v_t, y_bf);

  // output projection: [4096,2048] x [2048,2048]
  gemm_bf16<E_, E_><<<dim3(E_ / 256, (B_ * L_) / 128), 256, 0, stream>>>(
      y_bf, wpT, bp, out);
}
